// ViTMoE_7043746365558
// MI455X (gfx1250) — matrix operand order repaired and checked by reading, not compiled, not run on hardware
//
#include <hip/hip_runtime.h>
#include <hip/hip_bf16.h>

typedef __bf16 bf16_t;
typedef __attribute__((ext_vector_type(16))) __bf16 v16bf;
typedef __attribute__((ext_vector_type(8)))  __bf16 v8bf;
typedef __attribute__((ext_vector_type(8)))  float  v8f;

#define TOK    1576      // 8 * 197
#define EMB    768
#define NHEAD  12
#define HDIM   64
#define SEQ    197
#define SEQP   224       // padded seq (multiple of 32 for WMMA K)
#define NBATCH 8
#define NPATCH 196
#define HID1   3072
#define HIDB   6144
#define NEXPT  10
#define TOPK_  8
#define NCLS   1000

#define BM 128
#define BN 256
#define BK 32

__device__ __forceinline__ float gelu_f(float x) {
    return 0.5f * x * (1.0f + erff(x * 0.70710678118654752f));
}

// gfx1250 async global->LDS copy (ASYNCcnt-tracked, no VGPR round trip).
// VDST VGPR carries the LDS byte address (low 32 bits of the generic pointer).
__device__ __forceinline__ void async_ld_b128(void* lds, const void* gptr) {
    unsigned int l = (unsigned int)(unsigned long long)lds;
    asm volatile("global_load_async_to_lds_b128 %0, %1, off"
                 :: "v"(l), "v"(gptr) : "memory");
}
__device__ __forceinline__ void wait_async0() {
    asm volatile("s_wait_asynccnt 0x0" ::: "memory");
}

// ---------------------------------------------------------------------------
// Generic bf16 WMMA GEMM:  out = act(A[M,K] * Bt[N,K]^T + bias) (+ addsrc)
// Double-buffered LDS staging via async global->LDS; 16 WMMAs per wave per
// K-step.  C (fp32) and Cbf (bf16) outputs optional.  Batched via blockIdx.z.
// ---------------------------------------------------------------------------
__global__ __launch_bounds__(256)
void k_gemm_bf16(const bf16_t* __restrict__ A, const bf16_t* __restrict__ Bt,
                 const float* __restrict__ bias, const float* __restrict__ addsrc,
                 float* __restrict__ C, bf16_t* __restrict__ Cbf,
                 int M, int N, int K, int act,
                 long sA, long sB, long sC)
{
    __shared__ alignas(16) bf16_t lA[2][BM][BK];
    __shared__ alignas(32) bf16_t lB[2][BN][BK];

    const int tid  = threadIdx.x;
    const int lane = tid & 31;
    const int wv   = tid >> 5;
    const int wm   = wv >> 2;      // 0..1  (wave row)
    const int wn   = wv & 3;       // 0..3  (wave col)
    const int m0   = blockIdx.y * BM;
    const int n0   = blockIdx.x * BN;
    const long zb  = blockIdx.z;

    A  += zb * sA;
    Bt += zb * sB;
    if (C)      C      += zb * sC;
    if (Cbf)    Cbf    += zb * sC;
    if (addsrc) addsrc += zb * sC;

    v8f acc[4][4];
    const v8f vzero = {0.f, 0.f, 0.f, 0.f, 0.f, 0.f, 0.f, 0.f};
#pragma unroll
    for (int i = 0; i < 4; ++i)
#pragma unroll
        for (int j = 0; j < 4; ++j) acc[i][j] = vzero;

    // Per-thread staging sources.  Out-of-range rows/cols are CLAMPED to a
    // valid row: the data they produce only reaches accumulator rows/cols
    // that the guarded epilogue never stores, and K is an exact multiple of
    // BK, so no zero-fill is needed on the async path.
    const int a_r = tid >> 1;            // 0..127
    const int a_c = (tid & 1) * 16;      // 0 or 16 (bf16 elems)
    const int b_r = tid;                 // 0..255
    int gr = m0 + a_r; if (gr >= M) gr = M - 1;
    int gn = n0 + b_r; if (gn >= N) gn = N - 1;
    const bf16_t* gA = A  + (long)gr * K + a_c;
    const bf16_t* gB = Bt + (long)gn * K;

    const int nk = K / BK;

    // stage tile k0 into buffer buf (6 async b128 per thread = 24 KB/block)
    auto stage = [&](int buf, int k0) {
        async_ld_b128(&lA[buf][a_r][a_c],     gA + k0);
        async_ld_b128(&lA[buf][a_r][a_c + 8], gA + k0 + 8);
#pragma unroll
        for (int j = 0; j < 4; ++j)
            async_ld_b128(&lB[buf][b_r][j * 8], gB + k0 + j * 8);
    };

    stage(0, 0);
    wait_async0();
    __syncthreads();

    for (int ks = 0; ks < nk; ++ks) {
        const int cur = ks & 1;
        if (ks + 1 < nk) stage(cur ^ 1, (ks + 1) * BK);   // overlap with WMMA

        // ---- fragment loads: contiguous k = 16*(lane/16) + s for A and B ----
        v16bf af[4], bfr[4];
        {
            const int lr = lane & 15;
            const int kb = (lane >> 4) * 16;    // A half select
#pragma unroll
            for (int i = 0; i < 4; ++i) {
                const bf16_t* p = &lA[cur][wm * 64 + i * 16 + lr][0];
                v8bf lo = *(const v8bf*)(p + kb);
                v8bf hi = *(const v8bf*)(p + kb + 8);
                v16bf a;
#pragma unroll
                for (int e = 0; e < 8; ++e) { a[e] = lo[e]; a[e + 8] = hi[e]; }
                af[i] = a;
            }
            const int kb2 = (lane >> 4) * 16;   // B half select
#pragma unroll
            for (int j = 0; j < 4; ++j)
                bfr[j] = *(const v16bf*)(&lB[cur][wn * 64 + j * 16 + lr][kb2]);
        }

#pragma unroll
        for (int i = 0; i < 4; ++i)
#pragma unroll
            for (int j = 0; j < 4; ++j)
                acc[i][j] = __builtin_amdgcn_wmma_f32_16x16x32_bf16(
                    false, af[i], false, bfr[j], (short)0, acc[i][j], false, false);

        wait_async0();       // next tile resident in LDS
        __syncthreads();     // all waves done reading cur, all async done
    }

    // ---- epilogue ----
    const int lr   = lane & 15;
    const int rofs = (lane >> 4) * 8;
#pragma unroll
    for (int i = 0; i < 4; ++i) {
#pragma unroll
        for (int j = 0; j < 4; ++j) {
            const int colg = n0 + wn * 64 + j * 16 + lr;
#pragma unroll
            for (int r = 0; r < 8; ++r) {
                const int rowg = m0 + wm * 64 + i * 16 + rofs + r;
                if (rowg < M && colg < N) {
                    float v = acc[i][j][r];
                    if (bias) v += bias[colg];
                    if (act == 1) v = gelu_f(v);
                    const long off = (long)rowg * N + colg;
                    if (addsrc) v += addsrc[off];
                    if (C)   C[off]   = v;
                    if (Cbf) Cbf[off] = (bf16_t)v;
                }
            }
        }
    }
}

// ---------------------------------------------------------------------------
// Elementwise / helper kernels
// ---------------------------------------------------------------------------
__global__ void k_cvt(const float* __restrict__ in, bf16_t* __restrict__ out, long n) {
    long idx = (long)blockIdx.x * 256 + threadIdx.x;
    if (idx < n) out[idx] = (bf16_t)in[idx];
}

// in (K x N) row-major fp32 -> out (N x K) row-major bf16.
// LDS-tiled 32x32 transpose: coalesced reads AND writes (~1.04 GB of fp32 expert
// weights stream through here, so this must run at the HBM roofline).
__global__ __launch_bounds__(256)
void k_cvt_t(const float* __restrict__ in, bf16_t* __restrict__ out, int K, int N) {
    __shared__ float t[32][33];
    const int tx = threadIdx.x & 31;
    const int ty = threadIdx.x >> 5;     // 0..7
    const long k0 = (long)blockIdx.y * 32;
    const long n0 = (long)blockIdx.x * 32;
#pragma unroll
    for (int r = 0; r < 32; r += 8) {
        long k = k0 + ty + r, n = n0 + tx;
        t[ty + r][tx] = (k < K && n < N) ? in[k * N + n] : 0.f;
    }
    __syncthreads();
#pragma unroll
    for (int r = 0; r < 32; r += 8) {
        long n = n0 + ty + r, k = k0 + tx;
        if (n < N && k < K) out[n * K + k] = (bf16_t)t[tx][ty + r];
    }
}

__global__ void k_im2col(const float* __restrict__ x, bf16_t* __restrict__ A) {
    long idx = (long)blockIdx.x * 256 + threadIdx.x;
    long total = (long)NBATCH * NPATCH * EMB;
    if (idx >= total) return;
    int  k   = (int)(idx % EMB);
    long row = idx / EMB;
    int  b   = (int)(row / NPATCH);
    int  p   = (int)(row % NPATCH);
    int  py = p / 14, px = p % 14;
    int  c = k / 256, r2 = k % 256, i = r2 / 16, j = r2 % 16;
    A[idx] = (bf16_t)x[(((long)(b * 3 + c) * 224) + (py * 16 + i)) * 224 + (px * 16 + j)];
}

__global__ void k_assemble_h(const float* __restrict__ pemb, const float* __restrict__ cls,
                             const float* __restrict__ pos, float* __restrict__ h) {
    long idx = (long)blockIdx.x * 256 + threadIdx.x;
    long total = (long)TOK * EMB;
    if (idx >= total) return;
    int  e = (int)(idx % EMB);
    long t = idx / EMB;
    int  b = (int)(t / SEQ);
    int  n = (int)(t % SEQ);
    float v = (n == 0) ? cls[e] : pemb[((long)(b * NPATCH + (n - 1))) * EMB + e];
    h[idx] = v + pos[(long)n * EMB + e];
    (void)b;
}

__global__ __launch_bounds__(256)
void k_layernorm(const float* __restrict__ x, const float* __restrict__ g,
                 const float* __restrict__ b, bf16_t* __restrict__ o, int cols) {
    __shared__ float red[256];
    const long row = blockIdx.x;
    const float* xr = x + row * cols;
    float s = 0.f;
    for (int c = threadIdx.x; c < cols; c += 256) s += xr[c];
    red[threadIdx.x] = s; __syncthreads();
    for (int st = 128; st > 0; st >>= 1) {
        if (threadIdx.x < st) red[threadIdx.x] += red[threadIdx.x + st];
        __syncthreads();
    }
    const float mean = red[0] / cols; __syncthreads();
    float v = 0.f;
    for (int c = threadIdx.x; c < cols; c += 256) { float d = xr[c] - mean; v += d * d; }
    red[threadIdx.x] = v; __syncthreads();
    for (int st = 128; st > 0; st >>= 1) {
        if (threadIdx.x < st) red[threadIdx.x] += red[threadIdx.x + st];
        __syncthreads();
    }
    const float rstd = rsqrtf(red[0] / cols + 1e-5f);
    for (int c = threadIdx.x; c < cols; c += 256)
        o[row * cols + c] = (bf16_t)((xr[c] - mean) * rstd * g[c] + b[c]);
}

__global__ void k_qkv_split(const float* __restrict__ qkv, bf16_t* __restrict__ qb,
                            bf16_t* __restrict__ kb, bf16_t* __restrict__ vbT) {
    long idx = (long)blockIdx.x * 256 + threadIdx.x;
    long total = (long)96 * SEQ * HDIM;
    if (idx >= total) return;
    int  d  = (int)(idx % HDIM);
    long r  = idx / HDIM;
    int  n  = (int)(r % SEQ);
    int  bh = (int)(r / SEQ);
    int  b  = bh / NHEAD, h = bh % NHEAD;
    const float* base = qkv + ((long)b * SEQ + n) * (3 * EMB) + h * HDIM + d;
    qb[((long)bh * SEQP + n) * HDIM + d]  = (bf16_t)base[0];
    kb[((long)bh * SEQP + n) * HDIM + d]  = (bf16_t)base[EMB];
    vbT[((long)bh * HDIM + d) * SEQP + n] = (bf16_t)base[2 * EMB];
}

__global__ __launch_bounds__(256)
void k_softmax(const float* __restrict__ att, bf16_t* __restrict__ attb) {
    __shared__ float red[256];
    const long r   = blockIdx.x;            // 0 .. 96*197-1
    const long bh  = r / SEQ;
    const int  row = (int)(r % SEQ);
    const float* a = att + (bh * SEQP + row) * SEQP;
    bf16_t* ob = attb + (bh * SEQP + row) * SEQP;
    const float scale = 0.125f;             // 1/sqrt(64)
    float mx = -3.0e38f;
    for (int n = threadIdx.x; n < SEQ; n += 256) mx = fmaxf(mx, a[n] * scale);
    red[threadIdx.x] = mx; __syncthreads();
    for (int st = 128; st > 0; st >>= 1) {
        if (threadIdx.x < st) red[threadIdx.x] = fmaxf(red[threadIdx.x], red[threadIdx.x + st]);
        __syncthreads();
    }
    const float m = red[0]; __syncthreads();
    float s = 0.f;
    for (int n = threadIdx.x; n < SEQ; n += 256) s += expf(a[n] * scale - m);
    red[threadIdx.x] = s; __syncthreads();
    for (int st = 128; st > 0; st >>= 1) {
        if (threadIdx.x < st) red[threadIdx.x] += red[threadIdx.x + st];
        __syncthreads();
    }
    const float inv = 1.f / red[0];
    for (int n = threadIdx.x; n < SEQ; n += 256)
        ob[n] = (bf16_t)(expf(a[n] * scale - m) * inv);
}

__global__ void k_repack_o(const float* __restrict__ obh, bf16_t* __restrict__ ob) {
    long idx = (long)blockIdx.x * 256 + threadIdx.x;
    long total = (long)TOK * EMB;
    if (idx >= total) return;
    int  e = (int)(idx % EMB);
    long t = idx / EMB;
    int  b = (int)(t / SEQ), n = (int)(t % SEQ);
    int  h = e / HDIM, d = e % HDIM;
    ob[idx] = (bf16_t)obh[(((long)(b * NHEAD + h)) * SEQP + n) * HDIM + d];
}

__global__ void k_topk(const float* __restrict__ scores, float* __restrict__ comb) {
    int t = blockIdx.x * 256 + threadIdx.x;
    if (t >= TOK) return;
    float s[NEXPT]; bool used[NEXPT];
    for (int i = 0; i < NEXPT; ++i) { s[i] = scores[(long)t * NEXPT + i]; used[i] = false; }
    float vals[TOPK_]; int idxs[TOPK_];
    for (int k = 0; k < TOPK_; ++k) {
        float best = -3.0e38f; int bi = 0;
        for (int i = 0; i < NEXPT; ++i)
            if (!used[i] && s[i] > best) { best = s[i]; bi = i; }
        used[bi] = true; vals[k] = best; idxs[k] = bi;
    }
    const float m = vals[0];
    float sum = 0.f;
    for (int k = 0; k < TOPK_; ++k) sum += expf(vals[k] - m);
    for (int i = 0; i < NEXPT; ++i) comb[(long)t * NEXPT + i] = 0.f;
    for (int k = 0; k < TOPK_; ++k)
        comb[(long)t * NEXPT + idxs[k]] = expf(vals[k] - m) / sum;
}

__global__ void k_moe_acc(const float* __restrict__ eo, const float* __restrict__ comb,
                          float* __restrict__ moe, int x) {
    long idx = (long)blockIdx.x * 256 + threadIdx.x;
    long total = (long)TOK * EMB;
    if (idx >= total) return;
    long t = idx / EMB;
    moe[idx] += comb[t * NEXPT + x] * eo[idx];
}

__global__ void k_gather_cls(const bf16_t* __restrict__ lnf, bf16_t* __restrict__ Ah) {
    int idx = blockIdx.x * 256 + threadIdx.x;
    if (idx >= NBATCH * EMB) return;
    int e = idx % EMB, b = idx / EMB;
    Ah[(long)b * EMB + e] = lnf[((long)b * SEQ) * EMB + e];
}

// ---------------------------------------------------------------------------
extern "C" void kernel_launch(void* const* d_in, const int* in_sizes, int n_in,
                              void* d_out, int out_size, void* d_ws, size_t ws_size,
                              hipStream_t stream) {
    (void)in_sizes; (void)n_in; (void)out_size; (void)ws_size;

    const float* x       = (const float*)d_in[0];
    const float* conv_w  = (const float*)d_in[1];
    const float* conv_b  = (const float*)d_in[2];
    const float* cls_tok = (const float*)d_in[3];
    const float* pos_emb = (const float*)d_in[4];
    const float* ln1_g   = (const float*)d_in[5];
    const float* ln1_b   = (const float*)d_in[6];
    const float* qkv_w   = (const float*)d_in[7];
    const float* qkv_b   = (const float*)d_in[8];
    const float* out_w   = (const float*)d_in[9];
    const float* out_b   = (const float*)d_in[10];
    const float* ln2_g   = (const float*)d_in[11];
    const float* ln2_b   = (const float*)d_in[12];
    const float* mlp_w1  = (const float*)d_in[13];
    const float* mlp_b1  = (const float*)d_in[14];
    const float* mlp_w2  = (const float*)d_in[15];
    const float* mlp_b2  = (const float*)d_in[16];
    const float* gate_w  = (const float*)d_in[17];
    const float* gate_b  = (const float*)d_in[18];
    const float* ew1     = (const float*)d_in[19];
    const float* eb1     = (const float*)d_in[20];
    const float* ew2     = (const float*)d_in[21];
    const float* eb2     = (const float*)d_in[22];
    const float* ew3     = (const float*)d_in[23];
    const float* eb3     = (const float*)d_in[24];
    const float* lnf_g   = (const float*)d_in[25];
    const float* lnf_b   = (const float*)d_in[26];
    const float* head_w  = (const float*)d_in[27];
    const float* head_b  = (const float*)d_in[28];

    // -------- workspace layout (bump allocator, phase-overlapped) --------
    size_t off = 0;
    auto alloc = [&](size_t bytes) -> char* {
        char* p = (char*)d_ws + off;
        off = (off + bytes + 255) & ~(size_t)255;
        return p;
    };
    // persistent
    float*  h      = (float*) alloc((size_t)TOK * EMB * 4);
    bf16_t* h_bf   = (bf16_t*)alloc((size_t)TOK * EMB * 2);
    bf16_t* xn_bf  = (bf16_t*)alloc((size_t)TOK * EMB * 2);
    bf16_t* m_bf   = (bf16_t*)alloc((size_t)TOK * HID1 * 2);
    float*  eo     = (float*) alloc((size_t)TOK * EMB * 4);
    float*  moe    = (float*) alloc((size_t)TOK * EMB * 4);
    float*  scores = (float*) alloc((size_t)TOK * NEXPT * 4);
    float*  comb   = (float*) alloc((size_t)TOK * NEXPT * 4);
    bf16_t* lnfb   = (bf16_t*)alloc((size_t)TOK * EMB * 2);
    bf16_t* Ahead  = (bf16_t*)alloc((size_t)16 * EMB * 2);
    bf16_t* convwt = (bf16_t*)alloc((size_t)EMB * EMB * 2);
    bf16_t* qkvwt  = (bf16_t*)alloc((size_t)3 * EMB * EMB * 2);
    bf16_t* outwt  = (bf16_t*)alloc((size_t)EMB * EMB * 2);
    bf16_t* w1t    = (bf16_t*)alloc((size_t)HID1 * EMB * 2);
    bf16_t* w2t    = (bf16_t*)alloc((size_t)EMB * HID1 * 2);
    bf16_t* gatewt = (bf16_t*)alloc((size_t)NEXPT * EMB * 2);
    bf16_t* headwt = (bf16_t*)alloc((size_t)NCLS * EMB * 2);
    const size_t scratch_base = off;
    // phase 1: patch embed + attention
    bf16_t* peA  = (bf16_t*)alloc((size_t)NBATCH * NPATCH * EMB * 2);
    float*  pemb = (float*) alloc((size_t)NBATCH * NPATCH * EMB * 4);
    float*  qkvf = (float*) alloc((size_t)TOK * 3 * EMB * 4);
    bf16_t* qb   = (bf16_t*)alloc((size_t)96 * SEQP * HDIM * 2);
    bf16_t* kbm  = (bf16_t*)alloc((size_t)96 * SEQP * HDIM * 2);
    bf16_t* vbT  = (bf16_t*)alloc((size_t)96 * HDIM * SEQP * 2);
    float*  attf = (float*) alloc((size_t)96 * SEQP * SEQP * 4);
    bf16_t* attb = (bf16_t*)alloc((size_t)96 * SEQP * SEQP * 2);
    float*  obh  = (float*) alloc((size_t)96 * SEQP * HDIM * 4);
    bf16_t* o_bf = (bf16_t*)alloc((size_t)TOK * EMB * 2);
    const size_t end1 = off;
    off = scratch_base;
    // phase 2: MoE (overlaps phase-1 scratch)
    bf16_t* e1t = (bf16_t*)alloc((size_t)HID1 * EMB * 2);
    bf16_t* e2t = (bf16_t*)alloc((size_t)HIDB * HID1 * 2);
    bf16_t* e3t = (bf16_t*)alloc((size_t)EMB * HIDB * 2);
    bf16_t* e1a = (bf16_t*)alloc((size_t)TOK * HID1 * 2);
    bf16_t* e2a = (bf16_t*)alloc((size_t)TOK * HIDB * 2);
    const size_t end2 = off;
    const size_t ws_used = end1 > end2 ? end1 : end2;

    hipMemsetAsync(d_ws, 0, ws_used, stream);

    auto ggrid = [](int M, int N, int batch) {
        return dim3((unsigned)((N + BN - 1) / BN), (unsigned)((M + BM - 1) / BM), (unsigned)batch);
    };
    auto egrid = [](long n) { return dim3((unsigned)((n + 255) / 256)); };
    auto tgrid = [](int K, int N) {
        return dim3((unsigned)((N + 31) / 32), (unsigned)((K + 31) / 32));
    };

    // -------- weight conversion (bf16, transposed to N x K) --------
    k_cvt  <<<egrid((long)EMB * EMB), 256, 0, stream>>>(conv_w, convwt, (long)EMB * EMB); // already (N,K)
    k_cvt_t<<<tgrid(EMB, 3 * EMB), 256, 0, stream>>>(qkv_w, qkvwt, EMB, 3 * EMB);
    k_cvt_t<<<tgrid(EMB, EMB), 256, 0, stream>>>(out_w, outwt, EMB, EMB);
    k_cvt_t<<<tgrid(EMB, HID1), 256, 0, stream>>>(mlp_w1, w1t, EMB, HID1);
    k_cvt_t<<<tgrid(HID1, EMB), 256, 0, stream>>>(mlp_w2, w2t, HID1, EMB);
    k_cvt_t<<<tgrid(EMB, NEXPT), 256, 0, stream>>>(gate_w, gatewt, EMB, NEXPT);
    k_cvt_t<<<tgrid(EMB, NCLS), 256, 0, stream>>>(head_w, headwt, EMB, NCLS);

    // -------- patch embedding --------
    const int MP = NBATCH * NPATCH;   // 1568
    k_im2col<<<egrid((long)MP * EMB), 256, 0, stream>>>(x, peA);
    k_gemm_bf16<<<ggrid(MP, EMB, 1), 256, 0, stream>>>(peA, convwt, conv_b, nullptr,
        pemb, nullptr, MP, EMB, EMB, 0, 0, 0, 0);
    k_assemble_h<<<egrid((long)TOK * EMB), 256, 0, stream>>>(pemb, cls_tok, pos_emb, h);

    // -------- attention block --------
    k_layernorm<<<TOK, 256, 0, stream>>>(h, ln1_g, ln1_b, xn_bf, EMB);
    k_gemm_bf16<<<ggrid(TOK, 3 * EMB, 1), 256, 0, stream>>>(xn_bf, qkvwt, qkv_b, nullptr,
        qkvf, nullptr, TOK, 3 * EMB, EMB, 0, 0, 0, 0);
    k_qkv_split<<<egrid((long)96 * SEQ * HDIM), 256, 0, stream>>>(qkvf, qb, kbm, vbT);
    // scores = q @ k^T   (96 batches)
    k_gemm_bf16<<<ggrid(SEQP, SEQP, 96), 256, 0, stream>>>(qb, kbm, nullptr, nullptr,
        attf, nullptr, SEQP, SEQP, HDIM, 0,
        (long)SEQP * HDIM, (long)SEQP * HDIM, (long)SEQP * SEQP);
    k_softmax<<<96 * SEQ, 256, 0, stream>>>(attf, attb);
    // o = att @ v
    k_gemm_bf16<<<ggrid(SEQP, HDIM, 96), 256, 0, stream>>>(attb, vbT, nullptr, nullptr,
        obh, nullptr, SEQP, HDIM, SEQP, 0,
        (long)SEQP * SEQP, (long)HDIM * SEQP, (long)SEQP * HDIM);
    k_repack_o<<<egrid((long)TOK * EMB), 256, 0, stream>>>(obh, o_bf);
    // h += o @ out_w + out_b
    k_gemm_bf16<<<ggrid(TOK, EMB, 1), 256, 0, stream>>>(o_bf, outwt, out_b, h,
        h, nullptr, TOK, EMB, EMB, 0, 0, 0, 0);

    // -------- MLP block --------
    k_layernorm<<<TOK, 256, 0, stream>>>(h, ln2_g, ln2_b, xn_bf, EMB);
    k_gemm_bf16<<<ggrid(TOK, HID1, 1), 256, 0, stream>>>(xn_bf, w1t, mlp_b1, nullptr,
        nullptr, m_bf, TOK, HID1, EMB, 1, 0, 0, 0);
    k_gemm_bf16<<<ggrid(TOK, EMB, 1), 256, 0, stream>>>(m_bf, w2t, mlp_b2, h,
        h, h_bf, TOK, EMB, HID1, 0, 0, 0, 0);

    // -------- MoE --------
    k_gemm_bf16<<<ggrid(TOK, NEXPT, 1), 256, 0, stream>>>(h_bf, gatewt, gate_b, nullptr,
        scores, nullptr, TOK, NEXPT, EMB, 0, 0, 0, 0);
    k_topk<<<egrid(TOK), 256, 0, stream>>>(scores, comb);
    for (int xp = 0; xp < NEXPT; ++xp) {
        k_cvt_t<<<tgrid(EMB, HID1), 256, 0, stream>>>(ew1 + (long)xp * EMB * HID1, e1t, EMB, HID1);
        k_cvt_t<<<tgrid(HID1, HIDB), 256, 0, stream>>>(ew2 + (long)xp * HID1 * HIDB, e2t, HID1, HIDB);
        k_cvt_t<<<tgrid(HIDB, EMB), 256, 0, stream>>>(ew3 + (long)xp * HIDB * EMB, e3t, HIDB, EMB);
        k_gemm_bf16<<<ggrid(TOK, HID1, 1), 256, 0, stream>>>(h_bf, e1t, eb1 + (long)xp * HID1, nullptr,
            nullptr, e1a, TOK, HID1, EMB, 1, 0, 0, 0);
        k_gemm_bf16<<<ggrid(TOK, HIDB, 1), 256, 0, stream>>>(e1a, e2t, eb2 + (long)xp * HIDB, nullptr,
            nullptr, e2a, TOK, HIDB, HID1, 1, 0, 0, 0);
        k_gemm_bf16<<<ggrid(TOK, EMB, 1), 256, 0, stream>>>(e2a, e3t, eb3 + (long)xp * EMB, nullptr,
            eo, nullptr, TOK, EMB, HIDB, 0, 0, 0, 0);
        k_moe_acc<<<egrid((long)TOK * EMB), 256, 0, stream>>>(eo, comb, moe, xp);
    }

    // -------- final LN + head --------
    k_layernorm<<<TOK, 256, 0, stream>>>(moe, lnf_g, lnf_b, lnfb, EMB);
    k_gather_cls<<<egrid((long)NBATCH * EMB), 256, 0, stream>>>(lnfb, Ahead);
    k_gemm_bf16<<<ggrid(NBATCH, NCLS, 1), 256, 0, stream>>>(Ahead, headwt, head_b, nullptr,
        (float*)d_out, nullptr, NBATCH, NCLS, EMB, 0, 0, 0, 0);
}
